// MultiScaleAttentionPE_55250459296224
// MI455X (gfx1250) — compile-verified, matrix operand order checked by reading, not run on hardware
//
#include <hip/hip_runtime.h>

typedef __attribute__((ext_vector_type(16))) _Float16 v16h;
typedef __attribute__((ext_vector_type(8)))  _Float16 v8h;
typedef __attribute__((ext_vector_type(8)))  float    v8f;
typedef __attribute__((ext_vector_type(4)))  float    v4f;

namespace {
constexpr int kN0 = 65536, kN1 = 16384, kN2 = 4096, kN3 = 1024, kN4 = 256, kC = 256;
constexpr int kK  = 2 * kC;     // 512 contraction depth of the concat-GEMMs
constexpr int kBPad = 40;       // LDS B-tile row pitch in halves (80B: 20-dword stride)
}

// ---------------------------------------------------------------------------
// f = xyz @ W(3xC) + b         one block per point, one thread per channel
// ---------------------------------------------------------------------------
__global__ __launch_bounds__(256) void k_linear3(const float* __restrict__ xyz,
                                                 const float* __restrict__ W,
                                                 const float* __restrict__ b,
                                                 float* __restrict__ out) {
  const int r = blockIdx.x, c = threadIdx.x;
  const float x = xyz[r * 3 + 0], y = xyz[r * 3 + 1], z = xyz[r * 3 + 2];
  out[(size_t)r * kC + c] = b[c] + x * W[c] + y * W[kC + c] + z * W[2 * kC + c];
}

// cls4[c] = max over first N4 rows of f0
__global__ __launch_bounds__(256) void k_colmax(const float* __restrict__ f,
                                                float* __restrict__ out) {
  const int c = threadIdx.x;
  float m = -3.4e38f;
  for (int i = 0; i < kN4; ++i) m = fmaxf(m, f[(size_t)i * kC + c]);
  out[c] = m;
}

// gathered coordinates: out[i] = xyz0[idx[i]]
__global__ void k_gather3(const float* __restrict__ xyz0, const int* __restrict__ idx,
                          float* __restrict__ out, int M) {
  const int i = blockIdx.x * blockDim.x + threadIdx.x;
  if (i >= M) return;
  const int g = idx[i];
  out[i * 3 + 0] = xyz0[g * 3 + 0];
  out[i * 3 + 1] = xyz0[g * 3 + 1];
  out[i * 3 + 2] = xyz0[g * 3 + 2];
}

// P (512x256 f32, K-major) -> Pt (256x512 f16, N-major) so B tiles are
// contiguous 64B runs along K (async-copy friendly).
__global__ __launch_bounds__(256) void k_convP(const float* __restrict__ P,
                                               _Float16* __restrict__ Pt) {
  const int i = blockIdx.x * 256 + threadIdx.x;   // over 512*256
  const int k = i >> 8, n = i & 255;
  Pt[(size_t)n * kK + k] = (_Float16)P[i];
}

// f4_nei = cls4 + xyz4 @ W4 + b4
__global__ __launch_bounds__(256) void k_nei4(const float* __restrict__ cls4,
                                              const float* __restrict__ xyz4,
                                              const float* __restrict__ W,
                                              const float* __restrict__ b,
                                              float* __restrict__ out) {
  const int r = blockIdx.x, c = threadIdx.x;
  const float x = xyz4[r * 3 + 0], y = xyz4[r * 3 + 1], z = xyz4[r * 3 + 2];
  out[(size_t)r * kC + c] = cls4[c] + b[c] + x * W[c] + y * W[kC + c] + z * W[2 * kC + c];
}

// f_nei = pe_prev[k] + (qxyz - cxyz[k]) @ W + b    one block per query row
__global__ __launch_bounds__(256) void k_nei(const float* __restrict__ pe_prev,
                                             const int* __restrict__ kidx,
                                             const float* __restrict__ qxyz,
                                             const float* __restrict__ cxyz,
                                             const float* __restrict__ W,
                                             const float* __restrict__ b,
                                             float* __restrict__ out) {
  const int r = blockIdx.x, c = threadIdx.x;
  const int kk = kidx[r];
  const float dx = qxyz[r * 3 + 0] - cxyz[kk * 3 + 0];
  const float dy = qxyz[r * 3 + 1] - cxyz[kk * 3 + 1];
  const float dz = qxyz[r * 3 + 2] - cxyz[kk * 3 + 2];
  out[(size_t)r * kC + c] = pe_prev[(size_t)kk * kC + c] + b[c] +
                            dx * W[c] + dy * W[kC + c] + dz * W[2 * kC + c];
}

// ---------------------------------------------------------------------------
// split-parallel brute-force 1-NN with packed-u64 atomic argmin merge.
// Key = (float_bits(dist) << 32) | idx : dist >= 0 so float bits are monotone;
// ties resolve to smallest idx == jnp.argmin first-min semantics. atomicMin is
// order-independent -> deterministic.
// ---------------------------------------------------------------------------
__global__ void k_init_u64(unsigned long long* __restrict__ p) {
  p[blockIdx.x * 256 + threadIdx.x] = ~0ull;
}

__global__ __launch_bounds__(256) void k_nn1_part(const float* __restrict__ q,
                                                  const float* __restrict__ cand,
                                                  int McPer,
                                                  unsigned long long* __restrict__ best) {
  __shared__ float cx[256], cy[256], cz[256];
  const int r = blockIdx.x * 256 + threadIdx.x;
  const int c0 = blockIdx.y * McPer;
  const float qx = q[r * 3 + 0], qy = q[r * 3 + 1], qz = q[r * 3 + 2];
  float bd = 3.4e38f;
  int bi = 0;
  for (int base = 0; base < McPer; base += 256) {
    __syncthreads();
    const int ci = c0 + base + threadIdx.x;
    cx[threadIdx.x] = cand[ci * 3 + 0];
    cy[threadIdx.x] = cand[ci * 3 + 1];
    cz[threadIdx.x] = cand[ci * 3 + 2];
    __syncthreads();
    for (int j = 0; j < 256; ++j) {
      const float dx = qx - cx[j], dy = qy - cy[j], dz = qz - cz[j];
      const float d = dx * dx + dy * dy + dz * dz;
      if (d < bd) { bd = d; bi = c0 + base + j; }
    }
  }
  const unsigned long long pk =
      ((unsigned long long)__float_as_uint(bd) << 32) | (unsigned int)bi;
  atomicMin(best + r, pk);
}

__global__ void k_nn1_fin(const unsigned long long* __restrict__ best,
                          int* __restrict__ kout) {
  const int i = blockIdx.x * 256 + threadIdx.x;
  kout[i] = (int)(best[i] & 0xFFFFFFFFull);
}

// ---------------------------------------------------------------------------
// out[M x 256] = concat(Anei[M x 256], Af[M x 256]) @ P(512x256) + pb
//
// 8 waves/block; wave w: rows 16*(w/2), col-half 128*(w%2); block = 64x256.
// Per 32-deep K step: the 32x256 B tile (16 KB) is streamed into LDS with
// global_load_async_to_lds_b128 (ASYNCcnt, double buffered) while WMMAs run
// on the previous tile; A fragments are software-pipelined from global and
// B fragments are register double-buffered so ds latency hides under WMMA.
// ---------------------------------------------------------------------------
__global__ __launch_bounds__(256) void k_gemm_concat(const float* __restrict__ Anei,
                                                     const float* __restrict__ Af,
                                                     const _Float16* __restrict__ Pt,
                                                     const float* __restrict__ pb,
                                                     float* __restrict__ out) {
  __shared__ __align__(16) _Float16 Bt[2][256 * kBPad];   // 2 x 20 KB

  const int tid   = threadIdx.x;
  const int lane  = tid & 31;
  const int wave  = tid >> 5;            // 0..7
  const int mSub  = wave >> 1;           // 0..3
  const int nHalf = wave & 1;            // 0..1
  const int lrow  = lane & 15;           // row (A) / col (B,C) within tile
  const int hi    = lane >> 4;           // 0: lanes 0-15, 1: lanes 16-31
  const int off   = hi << 3;             // K sub-offset per CDNA5 16-bit A/B layout

  const int mBase = blockIdx.x * 64 + mSub * 16;
  const int aRow  = mBase + lrow;
  const int nBase = nHalf * 128;

  // B staging: thread t owns column n=t, copies 64B (32 halves) per K step.
  const int bn = tid;
  const _Float16* browg = Pt + (size_t)bn * kK;
  auto issueB = [&](int buf, int kb) {
    const unsigned long long g = (unsigned long long)(uintptr_t)(browg + kb);
    const unsigned int l = (unsigned int)(uintptr_t)(&Bt[buf][bn * kBPad]);
    asm volatile(
        "global_load_async_to_lds_b128 %0, %1, off\n\t"
        "global_load_async_to_lds_b128 %0, %1, off offset:16\n\t"
        "global_load_async_to_lds_b128 %0, %1, off offset:32\n\t"
        "global_load_async_to_lds_b128 %0, %1, off offset:48"
        :: "v"(l), "v"(g) : "memory");
  };

  // A fragment source for a K step (concat: K<256 -> f_nei, K>=256 -> f0)
  auto aptr = [&](int kb) {
    return (kb < kC) ? (Anei + (size_t)aRow * kC + kb)
                     : (Af + (size_t)aRow * kC + (kb - kC));
  };

  v8f acc[8];
#pragma unroll
  for (int t = 0; t < 8; ++t) {
    const float bv = pb[nBase + t * 16 + lrow];
#pragma unroll
    for (int v = 0; v < 8; ++v) acc[t][v] = bv;
  }

  // prologue: stage B tile 0, load A regs for step 0
  issueB(0, 0);
  const float* s0 = aptr(0);
  v4f a0 = *(const v4f*)(s0 + off);
  v4f a1 = *(const v4f*)(s0 + off + 4);
  v4f a2 = *(const v4f*)(s0 + off + 16);
  v4f a3 = *(const v4f*)(s0 + off + 20);

#pragma unroll
  for (int s = 0; s < 16; ++s) {
    const int buf = s & 1;
    v4f n0, n1, n2, n3;
    if (s < 15) {
      issueB(buf ^ 1, (s + 1) * 32);            // stream next B tile into LDS
      const float* ns = aptr((s + 1) * 32);     // and next A regs (no wait yet)
      n0 = *(const v4f*)(ns + off);
      n1 = *(const v4f*)(ns + off + 4);
      n2 = *(const v4f*)(ns + off + 16);
      n3 = *(const v4f*)(ns + off + 20);
      asm volatile("s_wait_asynccnt 0x4" ::: "memory");  // current tile landed
    } else {
      asm volatile("s_wait_asynccnt 0x0" ::: "memory");
    }
    __syncthreads();

    // A fragment: lane holds row aRow, K = kb + {off..off+7, 16+off..16+off+7}
    v16h a;
#pragma unroll
    for (int i = 0; i < 4; ++i) {
      a[i]      = (_Float16)a0[i];
      a[4 + i]  = (_Float16)a1[i];
      a[8 + i]  = (_Float16)a2[i];
      a[12 + i] = (_Float16)a3[i];
    }

    // B fragments from LDS, register double-buffered (preload t+1 before wmma t)
    const _Float16* bp0 = &Bt[buf][(nBase + lrow) * kBPad + off];
    v8h bc0 = *(const v8h*)bp0;
    v8h bc1 = *(const v8h*)(bp0 + 16);
#pragma unroll
    for (int t = 0; t < 8; ++t) {
      v8h bn0, bn1;
      if (t < 7) {
        const _Float16* bp = &Bt[buf][(nBase + (t + 1) * 16 + lrow) * kBPad + off];
        bn0 = *(const v8h*)bp;
        bn1 = *(const v8h*)(bp + 16);
      }
      const v16h bfrag = __builtin_shufflevector(bc0, bc1, 0, 1, 2, 3, 4, 5, 6, 7,
                                                 8, 9, 10, 11, 12, 13, 14, 15);
      acc[t] = __builtin_amdgcn_wmma_f32_16x16x32_f16(
          /*neg_a=*/false, a, /*neg_b=*/false, bfrag,
          /*c_mod=*/(short)0, acc[t], /*reuse_a=*/false, /*reuse_b=*/false);
      if (t < 7) { bc0 = bn0; bc1 = bn1; }
    }
    __syncthreads();
    if (s < 15) { a0 = n0; a1 = n1; a2 = n2; a3 = n3; }
  }

  // C/D layout: VGPR v -> row mBase + v + 8*hi, column nBase + t*16 + lrow
  const int oRow0 = mBase + (hi << 3);
#pragma unroll
  for (int t = 0; t < 8; ++t) {
    const int col = nBase + t * 16 + lrow;
#pragma unroll
    for (int v = 0; v < 8; ++v)
      out[(size_t)(oRow0 + v) * kC + col] = acc[t][v];
  }
}

// ---------------------------------------------------------------------------
extern "C" void kernel_launch(void* const* d_in, const int* in_sizes, int n_in,
                              void* d_out, int out_size, void* d_ws, size_t ws_size,
                              hipStream_t stream) {
  (void)in_sizes; (void)n_in; (void)out_size; (void)ws_size;
  const float* xyz0 = (const float*)d_in[0];
  const int*   idx0 = (const int*)d_in[1];
  const int*   idx1 = (const int*)d_in[2];
  const int*   idx2 = (const int*)d_in[3];
  const int*   idx3 = (const int*)d_in[4];
  const int*   idx4 = (const int*)d_in[5];
  const float *W_all = (const float*)d_in[6],  *b_all = (const float*)d_in[7];
  const float *W4 = (const float*)d_in[8],  *b4 = (const float*)d_in[9];
  const float *W3 = (const float*)d_in[10], *b3 = (const float*)d_in[11];
  const float *W2 = (const float*)d_in[12], *b2 = (const float*)d_in[13];
  const float *W1 = (const float*)d_in[14], *b1 = (const float*)d_in[15];
  const float *W0 = (const float*)d_in[16], *b0 = (const float*)d_in[17];
  const float *P4 = (const float*)d_in[18], *pb4 = (const float*)d_in[19];
  const float *P3 = (const float*)d_in[20], *pb3 = (const float*)d_in[21];
  const float *P2 = (const float*)d_in[22], *pb2 = (const float*)d_in[23];
  const float *P1 = (const float*)d_in[24], *pb1 = (const float*)d_in[25];
  const float *P0 = (const float*)d_in[26], *pb0 = (const float*)d_in[27];

  // outputs (flat, in return order)
  float* out = (float*)d_out;
  float* pe4 = out;                       // 256   x 256
  float* pe3 = pe4 + (size_t)kN4 * kC;    // 1024  x 256
  float* pe2 = pe3 + (size_t)kN3 * kC;    // 4096  x 256
  float* pe1 = pe2 + (size_t)kN2 * kC;    // 16384 x 256
  float* pe0 = pe1 + (size_t)kN1 * kC;    // 65536 x 256

  // workspace carve (256B aligned slices); ~131 MB total
  char* p = (char*)d_ws;
  auto carve = [&p](size_t bytes) {
    char* r = p;
    p += (bytes + 255) & ~(size_t)255;
    return r;
  };
  float* f0    = (float*)carve((size_t)kN0 * kC * sizeof(float));
  float* fnei  = (float*)carve((size_t)kN0 * kC * sizeof(float));
  float* cls4  = (float*)carve(kC * sizeof(float));
  float* xyz1  = (float*)carve((size_t)kN1 * 3 * sizeof(float));
  float* xyz2  = (float*)carve((size_t)kN2 * 3 * sizeof(float));
  float* xyz3  = (float*)carve((size_t)kN3 * 3 * sizeof(float));
  float* xyz4  = (float*)carve((size_t)kN4 * 3 * sizeof(float));
  int*   k34   = (int*)carve(kN3 * sizeof(int));
  int*   k23   = (int*)carve(kN2 * sizeof(int));
  int*   k12   = (int*)carve(kN1 * sizeof(int));
  unsigned long long* bb34 = (unsigned long long*)carve(kN3 * sizeof(unsigned long long));
  unsigned long long* bb23 = (unsigned long long*)carve(kN2 * sizeof(unsigned long long));
  unsigned long long* bb12 = (unsigned long long*)carve(kN1 * sizeof(unsigned long long));
  _Float16* Pt4 = (_Float16*)carve((size_t)kK * kC * sizeof(_Float16));
  _Float16* Pt3 = (_Float16*)carve((size_t)kK * kC * sizeof(_Float16));
  _Float16* Pt2 = (_Float16*)carve((size_t)kK * kC * sizeof(_Float16));
  _Float16* Pt1 = (_Float16*)carve((size_t)kK * kC * sizeof(_Float16));
  _Float16* Pt0 = (_Float16*)carve((size_t)kK * kC * sizeof(_Float16));

  // ---- stage 0: shared preprocessing ----
  k_linear3<<<kN0, 256, 0, stream>>>(xyz0, W_all, b_all, f0);
  k_gather3<<<kN1 / 256, 256, 0, stream>>>(xyz0, idx1, xyz1, kN1);
  k_gather3<<<kN2 / 256, 256, 0, stream>>>(xyz0, idx2, xyz2, kN2);
  k_gather3<<<kN3 / 256, 256, 0, stream>>>(xyz0, idx3, xyz3, kN3);
  k_gather3<<<1, 256, 0, stream>>>(xyz0, idx4, xyz4, kN4);
  k_convP<<<kK, 256, 0, stream>>>(P4, Pt4);
  k_convP<<<kK, 256, 0, stream>>>(P3, Pt3);
  k_convP<<<kK, 256, 0, stream>>>(P2, Pt2);
  k_convP<<<kK, 256, 0, stream>>>(P1, Pt1);
  k_convP<<<kK, 256, 0, stream>>>(P0, Pt0);
  k_init_u64<<<kN3 / 256, 256, 0, stream>>>(bb34);
  k_init_u64<<<kN2 / 256, 256, 0, stream>>>(bb23);
  k_init_u64<<<kN1 / 256, 256, 0, stream>>>(bb12);

  // ---- level 4 ----
  k_colmax<<<1, 256, 0, stream>>>(f0, cls4);
  k_nei4<<<kN4, 256, 0, stream>>>(cls4, xyz4, W4, b4, fnei);
  k_gemm_concat<<<kN4 / 64, 256, 0, stream>>>(fnei, f0, Pt4, pb4, pe4);

  // ---- level 3 ----  (1024 queries over 256 candidates, 1 chunk)
  k_nn1_part<<<dim3(kN3 / 256, 1), 256, 0, stream>>>(xyz3, xyz4, kN4, bb34);
  k_nn1_fin<<<kN3 / 256, 256, 0, stream>>>(bb34, k34);
  k_nei<<<kN3, 256, 0, stream>>>(pe4, k34, xyz3, xyz4, W3, b3, fnei);
  k_gemm_concat<<<kN3 / 64, 256, 0, stream>>>(fnei, f0, Pt3, pb3, pe3);

  // ---- level 2 ----  (4096 queries over 1024 candidates, 4 chunks)
  k_nn1_part<<<dim3(kN2 / 256, 4), 256, 0, stream>>>(xyz2, xyz3, kN3 / 4, bb23);
  k_nn1_fin<<<kN2 / 256, 256, 0, stream>>>(bb23, k23);
  k_nei<<<kN2, 256, 0, stream>>>(pe3, k23, xyz2, xyz3, W2, b2, fnei);
  k_gemm_concat<<<kN2 / 64, 256, 0, stream>>>(fnei, f0, Pt2, pb2, pe2);

  // ---- level 1 ----  (16384 queries over 4096 candidates, 8 chunks)
  k_nn1_part<<<dim3(kN1 / 256, 8), 256, 0, stream>>>(xyz1, xyz2, kN2 / 8, bb12);
  k_nn1_fin<<<kN1 / 256, 256, 0, stream>>>(bb12, k12);
  k_nei<<<kN1, 256, 0, stream>>>(pe2, k12, xyz1, xyz2, W1, b1, fnei);
  k_gemm_concat<<<kN1 / 64, 256, 0, stream>>>(fnei, f0, Pt1, pb1, pe1);

  // ---- level 0 ----
  k_nei<<<kN0, 256, 0, stream>>>(pe1, idx0, xyz0, xyz1, W0, b0, fnei);
  k_gemm_concat<<<kN0 / 64, 256, 0, stream>>>(fnei, f0, Pt0, pb0, pe0);
}
